// Decoder_73452530696379
// MI455X (gfx1250) — matrix operand order repaired and checked by reading, not compiled, not run on hardware
//
#include <hip/hip_runtime.h>
#include <hip/hip_bf16.h>
#include <math.h>

// Problem constants (from reference): B=512, T=128, E=256, D=256, OUT=1
#define BB  512
#define TT  128
#define EE  256
#define DD  256
#define G4  1024   // 4*D
#define KHC 512    // 2*D

typedef __attribute__((ext_vector_type(16))) __bf16        v16bf;
typedef __attribute__((ext_vector_type(8)))  float         v8f;
typedef __attribute__((ext_vector_type(8)))  unsigned int  v8u;
typedef __attribute__((ext_vector_type(4)))  int           v4i;
typedef __attribute__((ext_vector_type(4)))  unsigned int  u32x4;
typedef __attribute__((ext_vector_type(8)))  int           i32x8;
typedef __attribute__((ext_vector_type(4)))  int           i32x4;

#if defined(__gfx1250__) && \
    __has_builtin(__builtin_amdgcn_global_load_async_to_lds_b128) && \
    __has_builtin(__builtin_amdgcn_s_wait_asynccnt)
#define HAVE_ASYNC 1
#else
#define HAVE_ASYNC 0
#endif

#if defined(__gfx1250__) && \
    __has_builtin(__builtin_amdgcn_tensor_load_to_lds) && \
    __has_builtin(__builtin_amdgcn_s_wait_tensorcnt)
#define HAVE_TDM 1
#else
#define HAVE_TDM 0
#endif

// ---------- scalar helpers ----------
static __device__ __forceinline__ unsigned short f2bf(float f) {
    unsigned int u = __float_as_uint(f);
    u += 0x7FFFu + ((u >> 16) & 1u);   // round-to-nearest-even
    return (unsigned short)(u >> 16);
}
static __device__ __forceinline__ float bf2f(unsigned short h) {
    return __uint_as_float(((unsigned int)h) << 16);
}
static __device__ __forceinline__ float fast_tanh(float x) {
#if __has_builtin(__builtin_amdgcn_tanhf)
    return __builtin_amdgcn_tanhf(x);      // gfx1250 V_TANH_F32 (confirmed present)
#else
    float e = __expf(2.0f * x);
    return (e - 1.0f) / (e + 1.0f);
#endif
}
static __device__ __forceinline__ float sigmoidf(float x) {
    return 1.0f / (1.0f + __expf(-x));
}
static __device__ __forceinline__ float wave_sum(float v) {
#pragma unroll
    for (int m = 16; m > 0; m >>= 1) v += __shfl_xor(v, m, 32);
    return v;
}
static __device__ __forceinline__ float wave_max(float v) {
#pragma unroll
    for (int m = 16; m > 0; m >>= 1) v = fmaxf(v, __shfl_xor(v, m, 32));
    return v;
}

// ---------- WMMA fragment loaders (wave32, 16x16x32 bf16) ----------
static __device__ __forceinline__ v16bf load_a16x32(const unsigned short* A, int ldk,
                                                    int lane, int koff) {
    const int m  = lane & 15;
    const int hi = lane >> 4;
    const unsigned int* q = reinterpret_cast<const unsigned int*>(A + m * ldk + koff);
    const int b0 = hi * 8;
    const int b1 = hi * 8 + 4;
    v8u r;
#pragma unroll
    for (int i = 0; i < 4; ++i) { r[i] = q[b0 + i]; r[4 + i] = q[b1 + i]; }
    return __builtin_bit_cast(v16bf, r);
}
static __device__ __forceinline__ v16bf load_b16T(const unsigned short* W, int ldk,
                                                  int lane, int kbase, int nbase) {
    const int n    = nbase + (lane & 15);
    const int half = lane >> 4;
    const uint4* q = reinterpret_cast<const uint4*>(W + (size_t)n * ldk + kbase + half * 16);
    uint4 a = q[0], b = q[1];
    v8u r;
    r[0] = a.x; r[1] = a.y; r[2] = a.z; r[3] = a.w;
    r[4] = b.x; r[5] = b.y; r[6] = b.z; r[7] = b.w;
    return __builtin_bit_cast(v16bf, r);
}

#if HAVE_TDM
// One TDM descriptor moves a whole 32 KB chunk: 2D tile, tile_dim0 = 4*256 = 1024
// contiguous bf16 per batch row, tile_dim1 = 16 rows, row stride = T*E elements.
// D# layout per CDNA5 ISA 8.3/8.4 (group0: count/lds/global/type; group1: dims).
static __device__ __forceinline__ void tdm_load_chunk(const unsigned short* gsrc,
                                                      unsigned int lds_off) {
    unsigned long long ga = (unsigned long long)gsrc;
    u32x4 g0;
    g0[0] = 1u;                                                  // count=1 valid descriptor
    g0[1] = lds_off;                                             // lds_addr [63:32]
    g0[2] = (unsigned int)(ga & 0xFFFFFFFFu);                    // global_addr[31:0]
    g0[3] = (unsigned int)((ga >> 32) & 0x1FFFFFFu) | (2u << 30);// global_addr[56:32] | type=2
    i32x8 g1;
    g1[0] = (int)(1u << 16);       // workgroup_mask=0 | data_size=1 (2 bytes)
    g1[1] = (int)(1024u << 16);    // tensor_dim0[15:0]=1024 into bits [63:48]
    g1[2] = (int)(16u << 16);      // tensor_dim0 hi=0 | tensor_dim1[15:0]=16
    g1[3] = (int)(1024u << 16);    // tensor_dim1 hi=0 | tile_dim0=1024
    g1[4] = 16;                    // tile_dim1=16 | tile_dim2=0
    g1[5] = (int)(TT * EE);        // tensor_dim0_stride[31:0] = 32768 elems
    g1[6] = 0;                     // stride0 hi | tensor_dim1_stride lo
    g1[7] = 0;
    i32x4 z = {0, 0, 0, 0};
#if __clang_major__ >= 23
    i32x8 z8 = {0, 0, 0, 0, 0, 0, 0, 0};
    __builtin_amdgcn_tensor_load_to_lds(g0, g1, z, z, z8, 0);
#else
    __builtin_amdgcn_tensor_load_to_lds(g0, g1, z, z, 0);
#endif
}
#endif

#if HAVE_ASYNC
// Per-lane async-to-LDS staging: 512 threads x 4 b128 = one 32 KB chunk (ASYNCcnt).
static __device__ __forceinline__ void stage_chunk_async(const unsigned short* src,
                                                         unsigned short* dstbuf,
                                                         int row0, int c, int tid) {
#pragma unroll
    for (int j = 0; j < 4; ++j) {
        int i   = tid + 512 * j;      // b128 unit index, 0..2047
        int r   = i >> 7;             // row (0..15)
        int rem = i & 127;            // 16B unit within row-chunk
        int tl  = rem >> 5;           // t_local (0..3)
        int eo  = (rem & 31) << 3;    // e offset in elems
        const unsigned short* g = src + (((size_t)(row0 + r) * TT + c * 4 + tl) * EE + eo);
        unsigned short* l = dstbuf + ((size_t)i << 3);
        __builtin_amdgcn_global_load_async_to_lds_b128((v4i*)g, (v4i*)l, 0, 0);
    }
}
#endif

// ---------- prep kernels ----------
__global__ void k_f32_to_bf16(const float* __restrict__ s, unsigned short* __restrict__ d, int n) {
    int i = blockIdx.x * blockDim.x + threadIdx.x;
    int stride = gridDim.x * blockDim.x;
    for (; i < n; i += stride) d[i] = f2bf(s[i]);
}

__global__ void k_transpose_bf16(const float* __restrict__ s, int srcLd, int rowOff, int K,
                                 unsigned short* __restrict__ d, int dLd, int N) {
    int i = blockIdx.x * blockDim.x + threadIdx.x;
    int total = N * K;
    int stride = gridDim.x * blockDim.x;
    for (; i < total; i += stride) {
        int n_ = i / K, k_ = i % K;
        d[(size_t)n_ * dLd + k_] = f2bf(s[(size_t)(rowOff + k_) * srcLd + n_]);
    }
}

__global__ void k_bias_sum(const float* __restrict__ a, const float* __restrict__ b,
                           float* __restrict__ d, int n) {
    int i = blockIdx.x * blockDim.x + threadIdx.x;
    if (i < n) d[i] = a[i] + b[i];
}

// ---------- enc_proj GEMM (WMMA, two independent accumulator chains) ----------
__global__ void __launch_bounds__(256)
k_encproj(const unsigned short* __restrict__ Xb,
          const unsigned short* __restrict__ WencT,
          unsigned short* __restrict__ encp) {
    const int tid  = threadIdx.x;
    const int wave = tid >> 5;
    const int lane = tid & 31;
    const size_t row0 = (size_t)blockIdx.x * 128 + (size_t)wave * 16;
    const unsigned short* A = Xb + row0 * EE;

    v16bf af[8];
#pragma unroll
    for (int k = 0; k < 8; ++k) af[k] = load_a16x32(A, EE, lane, k * 32);

    const int n  = lane & 15;
    const int hi = lane >> 4;
#pragma unroll 1
    for (int nt = 0; nt < 16; ++nt) {
        v8f acc0 = {}, acc1 = {};
#pragma unroll
        for (int k = 0; k < 8; k += 2) {
            v16bf b0 = load_b16T(WencT, EE, lane, k * 32, nt * 16);
            v16bf b1 = load_b16T(WencT, EE, lane, (k + 1) * 32, nt * 16);
            acc0 = __builtin_amdgcn_wmma_f32_16x16x32_bf16(false, af[k],     false, b0,
                                                           (short)0, acc0, false, false);
            acc1 = __builtin_amdgcn_wmma_f32_16x16x32_bf16(false, af[k + 1], false, b1,
                                                           (short)0, acc1, false, false);
        }
#pragma unroll
        for (int i = 0; i < 8; ++i) {
            int m = i + 8 * hi;
            encp[(row0 + m) * EE + nt * 16 + n] = f2bf(acc0[i] + acc1[i]);
        }
    }
}

// ---------- fused decoder scan ----------
__global__ void __launch_bounds__(512)
k_decoder(const unsigned short* __restrict__ Xb,      // [B][T][E] bf16
          const unsigned short* __restrict__ encp,    // [B][T][E] bf16
          const unsigned short* __restrict__ aW1hcT,  // [E][2D] bf16 (transposed)
          const unsigned short* __restrict__ WhhB,    // [4D][D] bf16
          const float* __restrict__ bsum,
          const float* __restrict__ ab1,
          const float* __restrict__ aW2,
          const float* __restrict__ ab2,
          const float* __restrict__ Wih,
          const float* __restrict__ fcW,
          const float* __restrict__ fcb,
          const float* __restrict__ yhist,
          const float* __restrict__ ffW,
          const float* __restrict__ ffb,
          float* __restrict__ out)
{
    __shared__ float          sh_c[16 * EE];          // 16 KB (fp32 cell state)
    __shared__ unsigned short sh_hc[16 * KHC];        // 16 KB  [h | c] bf16 (WMMA A source)
    __shared__ float          sh_whc[16 * EE];        // 16 KB
    __shared__ float          sh_scores[16 * TT];     //  8 KB
    __shared__ unsigned short sh_attn[16 * TT];       //  4 KB
    __shared__ float          sh_ctx[16 * EE];        // 16 KB
    __shared__ float          sh_yt[16];
#if HAVE_TDM || HAVE_ASYNC
    __shared__ unsigned short sh_stage[2][16 * 4 * EE]; // 2 x 32 KB staging
#endif

    const int tid  = threadIdx.x;
    const int wave = tid >> 5;     // 0..15
    const int lane = tid & 31;
    const int row0 = blockIdx.x * 16;

    for (int i = tid; i < 16 * EE; i += 512) { sh_c[i] = 0.f; sh_ctx[i] = 0.f; }
    for (int i = tid; i < 16 * KHC; i += 512) sh_hc[i] = 0;
    __syncthreads();

    const int fbase = lane * 8;
    float aw[8], fcc[8];
#pragma unroll
    for (int j = 0; j < 8; ++j) { aw[j] = aW2[fbase + j]; fcc[j] = fcW[fbase + j]; }
    const float ab2v = ab2[0];
    const float fcbv = fcb[0];
    const float fcy  = fcW[EE];

#if HAVE_TDM || HAVE_ASYNC
    // 32-bit LDS byte offsets of the staging buffers (low 32 bits of generic address).
    const unsigned int lds_off0 = (unsigned int)(unsigned long long)(void*)&sh_stage[0][0];
    const unsigned int lds_off1 = (unsigned int)(unsigned long long)(void*)&sh_stage[1][0];
#endif

    for (int t = 0; t < TT; ++t) {
        // ---- Phase A: Whc = hc @ aW1[:2D] (+ab1). One n-tile per wave, K=512, 2 chains.
        {
            const int nb = wave * 16;
            v8f acc0 = {}, acc1 = {};
#pragma unroll
            for (int k = 0; k < KHC; k += 64) {
                v16bf a0 = load_a16x32(sh_hc, KHC, lane, k);
                v16bf b0 = load_b16T(aW1hcT, KHC, lane, k, nb);
                v16bf a1 = load_a16x32(sh_hc, KHC, lane, k + 32);
                v16bf b1 = load_b16T(aW1hcT, KHC, lane, k + 32, nb);
                acc0 = __builtin_amdgcn_wmma_f32_16x16x32_bf16(false, a0, false, b0,
                                                               (short)0, acc0, false, false);
                acc1 = __builtin_amdgcn_wmma_f32_16x16x32_bf16(false, a1, false, b1,
                                                               (short)0, acc1, false, false);
            }
            const int n = lane & 15, hi = lane >> 4;
            const float ab1v = ab1[nb + n];
#pragma unroll
            for (int i = 0; i < 8; ++i) {
                int m = i + 8 * hi;
                sh_whc[m * EE + nb + n] = acc0[i] + acc1[i] + ab1v;
            }
        }
        __syncthreads();

        // ---- Phase B: scores[r][tt] = sum_f tanh(encp + whc) * aW2[f].  Wave r = row.
        {
            const int r = wave;
            float wf[8];
#pragma unroll
            for (int j = 0; j < 8; ++j) wf[j] = sh_whc[r * EE + fbase + j];
#if HAVE_TDM
            if (tid == 0)
                tdm_load_chunk(encp + ((size_t)row0 * TT + 0) * EE, lds_off0);
#pragma unroll 1
            for (int c = 0; c < 32; ++c) {
                const int b = c & 1;
                if (tid == 0) {
                    if (c + 1 < 32) {
                        tdm_load_chunk(encp + ((size_t)row0 * TT + (c + 1) * 4) * EE,
                                       b ? lds_off0 : lds_off1);
                        __builtin_amdgcn_s_wait_tensorcnt(1);   // chunk c landed
                    } else {
                        __builtin_amdgcn_s_wait_tensorcnt(0);
                    }
                }
                __syncthreads();
#pragma unroll
                for (int tl = 0; tl < 4; ++tl) {
                    const uint4 v = *reinterpret_cast<const uint4*>(
                        &sh_stage[b][(r * 4 + tl) * EE + fbase]);
                    unsigned int w[4] = {v.x, v.y, v.z, v.w};
                    float s = 0.f;
#pragma unroll
                    for (int j = 0; j < 4; ++j) {
                        float z0 = bf2f((unsigned short)(w[j] & 0xffffu)) + wf[2 * j];
                        float z1 = bf2f((unsigned short)(w[j] >> 16)) + wf[2 * j + 1];
                        s += fast_tanh(z0) * aw[2 * j] + fast_tanh(z1) * aw[2 * j + 1];
                    }
                    s = wave_sum(s);
                    if (lane == 0) sh_scores[r * TT + c * 4 + tl] = s + ab2v;
                }
                __syncthreads();
            }
#else
            const unsigned short* eprow = encp + ((size_t)(row0 + r) * TT) * EE + fbase;
            for (int tt = 0; tt < TT; ++tt) {
                if (tt + 8 < TT) __builtin_prefetch(eprow + (size_t)(tt + 8) * EE, 0, 1);
                const uint4 v = *reinterpret_cast<const uint4*>(eprow + (size_t)tt * EE);
                unsigned int w[4] = {v.x, v.y, v.z, v.w};
                float s = 0.f;
#pragma unroll
                for (int j = 0; j < 4; ++j) {
                    float z0 = bf2f((unsigned short)(w[j] & 0xffffu)) + wf[2 * j];
                    float z1 = bf2f((unsigned short)(w[j] >> 16)) + wf[2 * j + 1];
                    s += fast_tanh(z0) * aw[2 * j] + fast_tanh(z1) * aw[2 * j + 1];
                }
                s = wave_sum(s);
                if (lane == 0) sh_scores[r * TT + tt] = s + ab2v;
            }
#endif
        }
        __syncthreads();

        // ---- Phase C: softmax over T per row.
        {
            const int r = wave;
            float v0[4];
#pragma unroll
            for (int j = 0; j < 4; ++j) v0[j] = sh_scores[r * TT + lane + 32 * j];
            float mx = fmaxf(fmaxf(v0[0], v0[1]), fmaxf(v0[2], v0[3]));
            mx = wave_max(mx);
            float e[4], s = 0.f;
#pragma unroll
            for (int j = 0; j < 4; ++j) { e[j] = __expf(v0[j] - mx); s += e[j]; }
            s = wave_sum(s);
            float inv = 1.0f / s;
#pragma unroll
            for (int j = 0; j < 4; ++j) sh_attn[r * TT + lane + 32 * j] = f2bf(e[j] * inv);
        }
        __syncthreads();

        // ---- Phase D: ctx[r][e] = sum_t attn[r][t] * X[r][t][e].  Wave r = row.
        {
            const int r = wave;
            float acc[8] = {};
#if HAVE_ASYNC
            stage_chunk_async(Xb, &sh_stage[0][0], row0, 0, tid);
#pragma unroll 1
            for (int c = 0; c < 32; ++c) {
                const int b = c & 1;
                if (c + 1 < 32) {
                    stage_chunk_async(Xb, &sh_stage[1 - b][0], row0, c + 1, tid);
                    __builtin_amdgcn_s_wait_asynccnt(4);
                } else {
                    __builtin_amdgcn_s_wait_asynccnt(0);
                }
                __syncthreads();
#pragma unroll
                for (int tl = 0; tl < 4; ++tl) {
                    float a = bf2f(sh_attn[r * TT + c * 4 + tl]);
                    const uint4 v = *reinterpret_cast<const uint4*>(
                        &sh_stage[b][(r * 4 + tl) * EE + fbase]);
                    unsigned int w[4] = {v.x, v.y, v.z, v.w};
#pragma unroll
                    for (int j = 0; j < 4; ++j) {
                        acc[2 * j]     += a * bf2f((unsigned short)(w[j] & 0xffffu));
                        acc[2 * j + 1] += a * bf2f((unsigned short)(w[j] >> 16));
                    }
                }
                __syncthreads();
            }
#else
            const unsigned short* xrow = Xb + ((size_t)(row0 + r) * TT) * EE + fbase;
            for (int tt = 0; tt < TT; ++tt) {
                if (tt + 8 < TT) __builtin_prefetch(xrow + (size_t)(tt + 8) * EE, 0, 1);
                float a = bf2f(sh_attn[r * TT + tt]);
                const uint4 v = *reinterpret_cast<const uint4*>(xrow + (size_t)tt * EE);
                unsigned int w[4] = {v.x, v.y, v.z, v.w};
#pragma unroll
                for (int j = 0; j < 4; ++j) {
                    acc[2 * j]     += a * bf2f((unsigned short)(w[j] & 0xffffu));
                    acc[2 * j + 1] += a * bf2f((unsigned short)(w[j] >> 16));
                }
            }
#endif
#pragma unroll
            for (int j = 0; j < 8; ++j) sh_ctx[r * EE + fbase + j] = acc[j];
        }
        __syncthreads();

        // ---- Phase E: y_tilde[r] = fcW[:256]·ctx + fcW[256]*y_t + fcb.
        {
            const int r = wave;
            float s = 0.f;
#pragma unroll
            for (int j = 0; j < 8; ++j) s += fcc[j] * sh_ctx[r * EE + fbase + j];
            s = wave_sum(s);
            if (lane == 0) {
                float y = yhist[(size_t)(row0 + r) * TT + t];
                sh_yt[r] = s + fcy * y + fcbv;
            }
        }
        __syncthreads();

        // ---- Phase F: gates = h@Whh.T (+bias, +y_tilde*Wih), then LSTM elementwise.
        {
            const int nb = wave * 16;
            v8f aI = {}, aF = {}, aG = {}, aO = {};
#pragma unroll
            for (int k = 0; k < DD; k += 32) {
                v16bf a = load_a16x32(sh_hc, KHC, lane, k);   // h part (cols 0..255)
                v16bf b;
                b  = load_b16T(WhhB, DD, lane, k, 0 * DD + nb);
                aI = __builtin_amdgcn_wmma_f32_16x16x32_bf16(false, a, false, b, (short)0, aI, false, false);
                b  = load_b16T(WhhB, DD, lane, k, 1 * DD + nb);
                aF = __builtin_amdgcn_wmma_f32_16x16x32_bf16(false, a, false, b, (short)0, aF, false, false);
                b  = load_b16T(WhhB, DD, lane, k, 2 * DD + nb);
                aG = __builtin_amdgcn_wmma_f32_16x16x32_bf16(false, a, false, b, (short)0, aG, false, false);
                b  = load_b16T(WhhB, DD, lane, k, 3 * DD + nb);
                aO = __builtin_amdgcn_wmma_f32_16x16x32_bf16(false, a, false, b, (short)0, aO, false, false);
            }
            __syncthreads();   // all waves done reading old h before anyone overwrites

            const int n = lane & 15, hi = lane >> 4;
            const int d = nb + n;
            const float bI = bsum[d],          bF = bsum[DD + d];
            const float bG = bsum[2 * DD + d], bO = bsum[3 * DD + d];
            const float wI = Wih[d],           wF = Wih[DD + d];
            const float wG = Wih[2 * DD + d],  wO = Wih[3 * DD + d];
#pragma unroll
            for (int i = 0; i < 8; ++i) {
                int m = i + 8 * hi;
                float yt = sh_yt[m];
                float gi = aI[i] + bI + yt * wI;
                float gf = aF[i] + bF + yt * wF;
                float gg = aG[i] + bG + yt * wG;
                float go = aO[i] + bO + yt * wO;
                float c_old = sh_c[m * EE + d];
                float c2 = sigmoidf(gf) * c_old + sigmoidf(gi) * fast_tanh(gg);
                float h2 = sigmoidf(go) * fast_tanh(c2);
                sh_c[m * EE + d] = c2;
                sh_hc[m * KHC + d]      = f2bf(h2);
                sh_hc[m * KHC + DD + d] = f2bf(c2);
            }
        }
        __syncthreads();
    }

    // ---- Final: out[row] = ffW[:256]·h + ffW[256:]·ctx + ffb
    {
        const int r = wave;
        float s = 0.f;
#pragma unroll
        for (int j = 0; j < 8; ++j) {
            s += ffW[fbase + j]      * bf2f(sh_hc[r * KHC + fbase + j]);
            s += ffW[EE + fbase + j] * sh_ctx[r * EE + fbase + j];
        }
        s = wave_sum(s);
        if (lane == 0) out[row0 + r] = s + ffb[0];
    }
}

// ---------- host launcher ----------
extern "C" void kernel_launch(void* const* d_in, const int* in_sizes, int n_in,
                              void* d_out, int out_size, void* d_ws, size_t ws_size,
                              hipStream_t stream) {
    (void)in_sizes; (void)n_in; (void)out_size; (void)ws_size;

    const float* input_encoded = (const float*)d_in[0];
    const float* y_history     = (const float*)d_in[1];
    const float* aW1 = (const float*)d_in[2];
    const float* ab1 = (const float*)d_in[3];
    const float* aW2 = (const float*)d_in[4];
    const float* ab2 = (const float*)d_in[5];
    const float* Wih = (const float*)d_in[6];
    const float* Whh = (const float*)d_in[7];
    const float* bih = (const float*)d_in[8];
    const float* bhh = (const float*)d_in[9];
    const float* fcW = (const float*)d_in[10];
    const float* fcb = (const float*)d_in[11];
    const float* ffW = (const float*)d_in[12];
    const float* ffb = (const float*)d_in[13];
    float* out = (float*)d_out;

    char* ws = (char*)d_ws;
    size_t off = 0;
    auto alloc = [&](size_t bytes) -> char* {
        char* p = ws + off;
        off += (bytes + 255) & ~(size_t)255;
        return p;
    };
    unsigned short* Xb      = (unsigned short*)alloc((size_t)BB * TT * EE * 2);
    unsigned short* encp    = (unsigned short*)alloc((size_t)BB * TT * EE * 2);
    unsigned short* aW1hcT  = (unsigned short*)alloc((size_t)EE * KHC * 2);
    unsigned short* aW1encT = (unsigned short*)alloc((size_t)EE * EE * 2);
    unsigned short* WhhB    = (unsigned short*)alloc((size_t)G4 * DD * 2);
    float*          bsum    = (float*)alloc((size_t)G4 * 4);

    int n;
    n = BB * TT * EE;
    k_f32_to_bf16<<<4096, 256, 0, stream>>>(input_encoded, Xb, n);
    n = G4 * DD;
    k_f32_to_bf16<<<(n + 255) / 256, 256, 0, stream>>>(Whh, WhhB, n);
    k_transpose_bf16<<<(EE * KHC + 255) / 256, 256, 0, stream>>>(aW1, EE, 0,   KHC, aW1hcT, KHC, EE);
    k_transpose_bf16<<<(EE * EE  + 255) / 256, 256, 0, stream>>>(aW1, EE, KHC, EE,  aW1encT, EE, EE);
    k_bias_sum<<<(G4 + 255) / 256, 256, 0, stream>>>(bih, bhh, bsum, G4);

    k_encproj<<<(BB * TT) / 128, 256, 0, stream>>>(Xb, aW1encT, encp);

    k_decoder<<<BB / 16, 512, 0, stream>>>(Xb, encp, aW1hcT, WhhB, bsum,
                                           ab1, aW2, ab2, Wih, fcW, fcb,
                                           y_history, ffW, ffb, out);
}
